// EBTAttention_3264175145204
// MI455X (gfx1250) — matrix operand order repaired and checked by reading, not compiled, not run on hardware
//
#include <hip/hip_runtime.h>
#include <hip/hip_bf16.h>

#define DIM     2048
#define NH      16
#define HD      128
#define SEQ     2048
#define FULL    4096

typedef __bf16           v16bf  __attribute__((ext_vector_type(16)));
typedef float            v8f    __attribute__((ext_vector_type(8)));
typedef unsigned short   v8u16  __attribute__((ext_vector_type(8)));
typedef unsigned short   v16u16 __attribute__((ext_vector_type(16)));
typedef unsigned int     u32x4  __attribute__((ext_vector_type(4)));
typedef int              i32x4  __attribute__((ext_vector_type(4)));
typedef int              i32x8  __attribute__((ext_vector_type(8)));

#if __has_builtin(__builtin_amdgcn_tensor_load_to_lds) && \
    __has_builtin(__builtin_amdgcn_s_wait_tensorcnt)
#define USE_TDM 1
#else
#define USE_TDM 0
#endif

// ---------- bf16 helpers (manual RNE, bit-exact control) ----------
static __device__ __forceinline__ unsigned short f2bf(float f) {
    unsigned int u = __builtin_bit_cast(unsigned int, f);
    unsigned int r = u + 0x7FFFu + ((u >> 16) & 1u);
    return (unsigned short)(r >> 16);
}
static __device__ __forceinline__ float bf2f(unsigned short u) {
    unsigned int x = ((unsigned int)u) << 16;
    return __builtin_bit_cast(float, x);
}

// ---------- WMMA fragment builders (layouts per CDNA5 ISA 7.12.2) ----------
// A fragment (16x32 bf16): lane m=lane&15; elems 0..7 = K[half*8 .. +7],
// elems 8..15 = K[16+half*8 .. +7], where half = lane>>4.
static __device__ __forceinline__ v16bf frag_a(const unsigned short* row, int half) {
    v8u16 c0 = *(const v8u16*)(row + half * 8);
    v8u16 c1 = *(const v8u16*)(row + 16 + half * 8);
    v16u16 t; t.lo = c0; t.hi = c1;
    return __builtin_bit_cast(v16bf, t);
}
// B fragment (32x16 bf16): lane n=lane&15; elems 0..7 = K[half*8 .. +7],
// elems 8..15 = K[16+half*8 .. +7], where half = lane>>4.
static __device__ __forceinline__ v16bf frag_b(const unsigned short* row, int half) {
    v8u16 c0 = *(const v8u16*)(row + half * 8);
    v8u16 c1 = *(const v8u16*)(row + 16 + half * 8);
    v16u16 t; t.lo = c0; t.hi = c1;
    return __builtin_bit_cast(v16bf, t);
}
static __device__ __forceinline__ v8f wmma_bf16(v16bf a, v16bf b, v8f c) {
    return __builtin_amdgcn_wmma_f32_16x16x32_bf16(false, a, false, b, (short)0, c,
                                                   false, false);
}

#if USE_TDM
// ---------- Tensor Data Mover: 2D tile (bf16 elements) global -> LDS ----------
// D# layout per CDNA5 ISA 8.3/8.4. tile_w elements per row (row copied
// contiguously into LDS), tile_h rows, row stride = stride_elems.
static __device__ __forceinline__ void tdm_load_2d(unsigned lds_addr,
                                                   const void* gaddr,
                                                   unsigned tile_w, unsigned tile_h,
                                                   unsigned tensor_w, unsigned tensor_h,
                                                   unsigned stride_elems) {
    unsigned long long ga = (unsigned long long)gaddr;
    u32x4 g0;
    g0[0] = 1u;                                       // count=1, user descriptor
    g0[1] = lds_addr;                                 // bits 63:32  lds_addr
    g0[2] = (unsigned)(ga & 0xFFFFFFFFu);             // bits 95:64  global_addr lo
    g0[3] = (unsigned)((ga >> 32) & 0x01FFFFFFu)      // bits 120:96 global_addr hi
          | (2u << 30);                               // bits 127:126 type=2 (image)
    i32x8 g1;
    g1[0] = (int)(1u << 16);                          // data_size=1 (2 bytes/elem)
    g1[1] = (int)((tensor_w & 0xFFFFu) << 16);        // tensor_dim0[15:0]
    g1[2] = (int)((tensor_w >> 16) | ((tensor_h & 0xFFFFu) << 16)); // dim0 hi | dim1 lo
    g1[3] = (int)((tensor_h >> 16) | (tile_w << 16)); // dim1 hi | tile_dim0
    g1[4] = (int)(tile_h & 0xFFFFu);                  // tile_dim1 (tile_dim2=0)
    g1[5] = (int)stride_elems;                        // tensor_dim0_stride[31:0]
    g1[6] = 0;                                        // stride hi | dim1_stride lo
    g1[7] = 0;
    i32x4 z4 = {0, 0, 0, 0};
#if defined(__clang_major__) && (__clang_major__ >= 23)
    i32x8 z8 = {0, 0, 0, 0, 0, 0, 0, 0};
    __builtin_amdgcn_tensor_load_to_lds(g0, g1, z4, z4, z8, 0);
#else
    __builtin_amdgcn_tensor_load_to_lds(g0, g1, z4, z4, 0);
#endif
}
#endif  // USE_TDM

// ---------- fp32 -> bf16 convert ----------
__global__ __launch_bounds__(256) void cvt_kernel(const float* __restrict__ in,
                                                  unsigned short* __restrict__ out,
                                                  size_t n) {
    size_t i = (size_t)blockIdx.x * 256 + threadIdx.x;
    if (i < n) out[i] = f2bf(in[i]);
}

// ---------- GEMM: C = A (MxK, bf16 rowmajor) * B^T (B is NxK, bf16 rowmajor) ----------
// 128x128 block tile, 8 waves, 32x64 wave tile -> 8 WMMAs per wave per k-step.
// mode 0: C = float, row-major MxN.   mode 1: C = bf16, head-major [h][m][d] (n = h*128+d).
__global__ __launch_bounds__(256) void gemm_abT(const unsigned short* __restrict__ A,
                                                const unsigned short* __restrict__ B,
                                                void* __restrict__ C,
                                                int M, int N, int K, int mode) {
    __shared__ __align__(64) unsigned short As[128 * 32];
    __shared__ __align__(64) unsigned short Bs[128 * 32];

    const int m0   = blockIdx.y * 128;
    const int n0   = blockIdx.x * 128;
    const int t    = threadIdx.x;
    const int wave = t >> 5;
    const int lane = t & 31;
    const int half = lane >> 4;
    const int l15  = lane & 15;
    const int wr   = (wave & 3) * 32;   // wave row offset within 128x128 tile
    const int wc   = (wave >> 2) * 64;  // wave col offset

    v8f acc[2][4];
    #pragma unroll
    for (int i = 0; i < 2; ++i)
        #pragma unroll
        for (int j = 0; j < 4; ++j) acc[i][j] = (v8f){};

#if USE_TDM
    const unsigned lds_as = (unsigned)(unsigned long long)(void*)As;
    const unsigned lds_bs = (unsigned)(unsigned long long)(void*)Bs;
#else
    const int lrow = t >> 1;            // cooperative-load row (0..127)
    const int lcol = (t & 1) * 16;      // cooperative-load col {0,16}
#endif

    for (int k0 = 0; k0 < K; k0 += 32) {
#if USE_TDM
        if (wave == 0) {  // TDM: EXEC-independent async DMA, per-wave issue
            tdm_load_2d(lds_as, &A[(size_t)m0 * K + k0], 32, 128,
                        (unsigned)K, (unsigned)M, (unsigned)K);
            tdm_load_2d(lds_bs, &B[(size_t)n0 * K + k0], 32, 128,
                        (unsigned)K, (unsigned)N, (unsigned)K);
            __builtin_amdgcn_s_wait_tensorcnt(0);
        }
#else
        *(v8u16*)&As[lrow * 32 + lcol] =
            *(const v8u16*)&A[(size_t)(m0 + lrow) * K + k0 + lcol];
        *(v8u16*)&As[lrow * 32 + lcol + 8] =
            *(const v8u16*)&A[(size_t)(m0 + lrow) * K + k0 + lcol + 8];
        *(v8u16*)&Bs[lrow * 32 + lcol] =
            *(const v8u16*)&B[(size_t)(n0 + lrow) * K + k0 + lcol];
        *(v8u16*)&Bs[lrow * 32 + lcol + 8] =
            *(const v8u16*)&B[(size_t)(n0 + lrow) * K + k0 + lcol + 8];
        if (k0 + 32 < K) {
            __builtin_prefetch(&A[(size_t)(m0 + lrow) * K + k0 + 32 + lcol], 0, 1);
            __builtin_prefetch(&B[(size_t)(n0 + lrow) * K + k0 + 32 + lcol], 0, 1);
        }
#endif
        __syncthreads();

        v16bf a0 = frag_a(&As[(wr + l15) * 32], half);
        v16bf a1 = frag_a(&As[(wr + 16 + l15) * 32], half);
        v16bf b0 = frag_b(&Bs[(wc + l15) * 32], half);
        v16bf b1 = frag_b(&Bs[(wc + 16 + l15) * 32], half);
        v16bf b2 = frag_b(&Bs[(wc + 32 + l15) * 32], half);
        v16bf b3 = frag_b(&Bs[(wc + 48 + l15) * 32], half);
        acc[0][0] = wmma_bf16(a0, b0, acc[0][0]);
        acc[0][1] = wmma_bf16(a0, b1, acc[0][1]);
        acc[0][2] = wmma_bf16(a0, b2, acc[0][2]);
        acc[0][3] = wmma_bf16(a0, b3, acc[0][3]);
        acc[1][0] = wmma_bf16(a1, b0, acc[1][0]);
        acc[1][1] = wmma_bf16(a1, b1, acc[1][1]);
        acc[1][2] = wmma_bf16(a1, b2, acc[1][2]);
        acc[1][3] = wmma_bf16(a1, b3, acc[1][3]);
        __syncthreads();
    }

    // C layout: lane holds col n=lane&15; rows m = r + 8*half for vgpr r.
    #pragma unroll
    for (int i = 0; i < 2; ++i) {
        #pragma unroll
        for (int j = 0; j < 4; ++j) {
            #pragma unroll
            for (int r = 0; r < 8; ++r) {
                int m = m0 + wr + i * 16 + r + half * 8;
                int n = n0 + wc + j * 16 + l15;
                float v = acc[i][j][r];
                if (mode == 0) {
                    ((float*)C)[(size_t)m * N + n] = v;
                } else {
                    int hh = n >> 7, dd = n & 127;
                    ((unsigned short*)C)[((size_t)hh * FULL + m) * HD + dd] = f2bf(v);
                }
            }
        }
    }
}

// ---------- RoPE in-place on head-major q/k (bf16) ----------
__global__ __launch_bounds__(256) void rope_kernel(unsigned short* __restrict__ qb,
                                                   unsigned short* __restrict__ kb,
                                                   const float* __restrict__ cosT,
                                                   const float* __restrict__ sinT) {
    size_t idx = (size_t)blockIdx.x * 256 + threadIdx.x;  // over NH*FULL*(HD/2)
    if (idx >= (size_t)NH * FULL * (HD / 2)) return;
    int    j    = idx & 63;
    size_t hs   = idx >> 6;              // h*FULL + s
    int    s    = (int)(hs & (FULL - 1));
    int    frow = (s < SEQ) ? s : (s - SEQ + 1);
    float  c  = cosT[frow * 64 + j];
    float  sn = sinT[frow * 64 + j];
    size_t base = hs * HD + 2 * j;
    {
        float xr = bf2f(qb[base]), xi = bf2f(qb[base + 1]);
        qb[base]     = f2bf(xr * c - xi * sn);
        qb[base + 1] = f2bf(xr * sn + xi * c);
    }
    {
        float xr = bf2f(kb[base]), xi = bf2f(kb[base + 1]);
        kb[base]     = f2bf(xr * c - xi * sn);
        kb[base + 1] = f2bf(xr * sn + xi * c);
    }
}

// ---------- V transpose per head: vtb[h][d][s] = vb[h][s][d] ----------
__global__ __launch_bounds__(256) void vtrans_kernel(const unsigned short* __restrict__ vb,
                                                     unsigned short* __restrict__ vtb) {
    size_t idx = (size_t)blockIdx.x * 256 + threadIdx.x;  // over NH*FULL*HD
    if (idx >= (size_t)NH * FULL * HD) return;
    int    d  = idx & 127;
    size_t hs = idx >> 7;
    int    s  = (int)(hs & (FULL - 1));
    int    h  = (int)(hs >> 12);  // FULL = 4096 = 2^12
    vtb[((size_t)h * HD + d) * FULL + s] = vb[idx];
}

// ---------- Flash attention with EBT superdiagonal ----------
// grid: (FULL/64, NH), block: 128 (4 waves x 16 query rows each)
__global__ __launch_bounds__(128) void attn_kernel(const unsigned short* __restrict__ qb,
                                                   const unsigned short* __restrict__ kb,
                                                   const unsigned short* __restrict__ vb,
                                                   const unsigned short* __restrict__ vtb,
                                                   unsigned short* __restrict__ attn) {
    __shared__ __align__(64) unsigned short pbuf[4][16][32];
    __shared__ float sdot[4][16];

    const int h     = blockIdx.y;
    const int qtile = blockIdx.x;
    const int wave  = threadIdx.x >> 5;
    const int lane  = threadIdx.x & 31;
    const int half  = lane >> 4;
    const int l15   = lane & 15;

    const int  qbase      = qtile * 64 + wave * 16;   // this wave's first query row
    const bool is_p       = (qtile * 64) >= SEQ;      // uniform per block
    const int  limit_base = is_p ? (qbase - SEQ) : qbase;
    const int  kmax_block = (is_p ? (qtile * 64 - SEQ) : (qtile * 64)) + 63;
    const float inv_sqrt  = 0.08838834764831845f;     // 1/sqrt(128)

    const unsigned short* Qh = qb  + (size_t)h * FULL * HD;
    const unsigned short* Kh = kb  + (size_t)h * FULL * HD;
    const unsigned short* VT = vtb + (size_t)h * HD * FULL;

    // Preload Q A-fragments for the 4 d-steps (d = 0..127 in chunks of 32)
    v16bf qf[4];
    {
        const unsigned short* qr = Qh + (size_t)(qbase + l15) * HD;
        #pragma unroll
        for (int ds = 0; ds < 4; ++ds) qf[ds] = frag_a(qr + ds * 32, half);
    }

    v8f O[8];
    #pragma unroll
    for (int c = 0; c < 8; ++c) O[c] = (v8f){};
    float rmax[8], rsum[8];
    #pragma unroll
    for (int r = 0; r < 8; ++r) { rmax[r] = -1e30f; rsum[r] = 0.0f; }

    for (int k0 = 0; k0 <= kmax_block; k0 += 32) {
        // ---- scores S = Q * K^T over 32 keys (two 16x16 tiles) ----
        v8f s0 = {}, s1 = {};
        #pragma unroll
        for (int ds = 0; ds < 4; ++ds) {
            v16bf kf0 = frag_b(Kh + (size_t)(k0 + l15) * HD + ds * 32, half);
            v16bf kf1 = frag_b(Kh + (size_t)(k0 + 16 + l15) * HD + ds * 32, half);
            s0 = wmma_bf16(qf[ds], kf0, s0);
            s1 = wmma_bf16(qf[ds], kf1, s1);
        }

        // ---- mask + online softmax ----
        float p0[8], p1[8], corr[8];
        #pragma unroll
        for (int r = 0; r < 8; ++r) {
            int   row   = r + half * 8;
            int   limit = limit_base + row;
            int   key0  = k0 + l15;
            int   key1  = key0 + 16;
            float v0 = (key0 <= limit) ? s0[r] * inv_sqrt : -1e30f;
            float v1 = (key1 <= limit) ? s1[r] * inv_sqrt : -1e30f;
            float tm = fmaxf(v0, v1);
            #pragma unroll
            for (int m = 1; m < 16; m <<= 1) tm = fmaxf(tm, __shfl_xor(tm, m, 32));
            float nm = fmaxf(rmax[r], tm);
            corr[r]  = __expf(rmax[r] - nm);
            rmax[r]  = nm;
            p0[r] = __expf(v0 - nm);
            p1[r] = __expf(v1 - nm);
            float ts = p0[r] + p1[r];
            #pragma unroll
            for (int m = 1; m < 16; m <<= 1) ts += __shfl_xor(ts, m, 32);
            rsum[r] = rsum[r] * corr[r] + ts;
        }
        #pragma unroll
        for (int c = 0; c < 8; ++c)
            #pragma unroll
            for (int r = 0; r < 8; ++r) O[c][r] *= corr[r];

        // ---- bounce P through LDS: C-layout -> A-layout ----
        #pragma unroll
        for (int r = 0; r < 8; ++r) {
            int row = r + half * 8;
            pbuf[wave][row][l15]      = f2bf(p0[r]);
            pbuf[wave][row][l15 + 16] = f2bf(p1[r]);
        }
        __syncthreads();
        v16bf pf = frag_a(&pbuf[wave][l15][0], half);

        // ---- O += P * V  (B fragments from V^T: contiguous keys) ----
        #pragma unroll
        for (int c = 0; c < 8; ++c) {
            v16bf vf = frag_b(VT + (size_t)(c * 16 + l15) * FULL + k0, half);
            O[c] = wmma_bf16(pf, vf, O[c]);
        }
        __syncthreads();
    }

    // ---- EBT superdiagonal self term for p-half ----
    float pself[8];
    #pragma unroll
    for (int r = 0; r < 8; ++r) pself[r] = 0.0f;
    if (is_p) {
        int row = l15;
        int q   = qbase + row;  // = SEQ + r
        const unsigned short* qp = Qh + (size_t)q * HD + half * 64;
        const unsigned short* kp = Kh + (size_t)q * HD + half * 64;
        float dsum = 0.0f;
        #pragma unroll 8
        for (int i = 0; i < 64; ++i) dsum += bf2f(qp[i]) * bf2f(kp[i]);
        dsum += __shfl_xor(dsum, 16, 32);
        if (half == 0) sdot[wave][row] = dsum;
        __syncthreads();
        #pragma unroll
        for (int r = 0; r < 8; ++r) {
            int   rr = r + half * 8;
            float ss = sdot[wave][rr] * inv_sqrt;
            float nm = fmaxf(rmax[r], ss);
            float cr = __expf(rmax[r] - nm);
            pself[r] = __expf(ss - nm);
            rsum[r]  = rsum[r] * cr + pself[r];
            rmax[r]  = nm;
            #pragma unroll
            for (int c = 0; c < 8; ++c) O[c][r] *= cr;
        }
    }

    // ---- normalize + (p-half) v_p contribution + store ----
    #pragma unroll
    for (int c = 0; c < 8; ++c) {
        #pragma unroll
        for (int r = 0; r < 8; ++r) {
            int   rr  = r + half * 8;
            int   q   = qbase + rr;
            int   d   = c * 16 + l15;
            float inv = 1.0f / rsum[r];
            float val = O[c][r] * inv;
            if (is_p)
                val += pself[r] * inv * bf2f(vb[((size_t)h * FULL + q) * HD + d]);
            attn[(size_t)q * DIM + h * HD + d] = f2bf(val);
        }
    }
}

extern "C" void kernel_launch(void* const* d_in, const int* in_sizes, int n_in,
                              void* d_out, int out_size, void* d_ws, size_t ws_size,
                              hipStream_t stream) {
    const float* x    = (const float*)d_in[0];
    const float* fcos = (const float*)d_in[1];
    const float* fsin = (const float*)d_in[2];
    // d_in[3] = mask (unused; causal structure computed analytically)
    const float* wq = (const float*)d_in[4];
    const float* wk = (const float*)d_in[5];
    const float* wv = (const float*)d_in[6];
    const float* wo = (const float*)d_in[7];

    // Workspace carve-up (bf16 as ushort): 128 MB total
    unsigned short* xb   = (unsigned short*)d_ws;
    unsigned short* wqb  = xb   + (size_t)FULL * DIM;
    unsigned short* wkb  = wqb  + (size_t)DIM * DIM;
    unsigned short* wvb  = wkb  + (size_t)DIM * DIM;
    unsigned short* wob  = wvb  + (size_t)DIM * DIM;
    unsigned short* qb   = wob  + (size_t)DIM * DIM;   // head-major [h][s][d]
    unsigned short* kbuf = qb   + (size_t)NH * FULL * HD;
    unsigned short* vbuf = kbuf + (size_t)NH * FULL * HD;
    unsigned short* vtb  = vbuf + (size_t)NH * FULL * HD;  // [h][d][s]
    unsigned short* attn = vtb  + (size_t)NH * FULL * HD;  // [s][dim] bf16

    const size_t nx = (size_t)FULL * DIM;
    const size_t nw = (size_t)DIM * DIM;
    cvt_kernel<<<(unsigned)((nx + 255) / 256), 256, 0, stream>>>(x, xb, nx);
    cvt_kernel<<<(unsigned)((nw + 255) / 256), 256, 0, stream>>>(wq, wqb, nw);
    cvt_kernel<<<(unsigned)((nw + 255) / 256), 256, 0, stream>>>(wk, wkb, nw);
    cvt_kernel<<<(unsigned)((nw + 255) / 256), 256, 0, stream>>>(wv, wvb, nw);
    cvt_kernel<<<(unsigned)((nw + 255) / 256), 256, 0, stream>>>(wo, wob, nw);

    dim3 ggrid(DIM / 128, FULL / 128);
    gemm_abT<<<ggrid, 256, 0, stream>>>(xb, wqb, qb,   FULL, DIM, DIM, 1);
    gemm_abT<<<ggrid, 256, 0, stream>>>(xb, wkb, kbuf, FULL, DIM, DIM, 1);
    gemm_abT<<<ggrid, 256, 0, stream>>>(xb, wvb, vbuf, FULL, DIM, DIM, 1);

    const size_t nr = (size_t)NH * FULL * (HD / 2);
    rope_kernel<<<(unsigned)((nr + 255) / 256), 256, 0, stream>>>(qb, kbuf, fcos, fsin);

    const size_t nv = (size_t)NH * FULL * HD;
    vtrans_kernel<<<(unsigned)((nv + 255) / 256), 256, 0, stream>>>(vbuf, vtb);

    attn_kernel<<<dim3(FULL / 64, NH), 128, 0, stream>>>(qb, kbuf, vbuf, vtb, attn);

    gemm_abT<<<ggrid, 256, 0, stream>>>(attn, wob, d_out, FULL, DIM, DIM, 0);
}
